// BasicBlockBase_40604620816612
// MI455X (gfx1250) — matrix operand order repaired and checked by reading, not compiled, not run on hardware
//
#include <hip/hip_runtime.h>

#define C_DIM 64
#define EPSV 1e-5f
#define PBLK 240

typedef unsigned short u16;
typedef __bf16 bf16_t;
typedef bf16_t v16bf __attribute__((ext_vector_type(16)));
typedef float  v8f   __attribute__((ext_vector_type(8)));

__device__ __forceinline__ u16 f2bf(float f) {
  unsigned int u = __float_as_uint(f);
  u += 0x7FFFu + ((u >> 16) & 1u);   // round-to-nearest-even
  return (u16)(u >> 16);
}

union FragA { uint4 q[2]; v16bf v; };

// ---------------------------------------------------------------------------
// Prep: f32 -> bf16 feature table
// ---------------------------------------------------------------------------
__global__ __launch_bounds__(256) void cvt_bf16(const float* __restrict__ x,
                                                u16* __restrict__ t, long n) {
  long i = (long)blockIdx.x * blockDim.x + threadIdx.x;
  if (i < n) t[i] = f2bf(x[i]);
}

// ---------------------------------------------------------------------------
// Prep: swizzle W[k][c][d] (f32) into per-lane WMMA B fragments (bf16).
// Fragment id = (k*2 + t)*4 + j ; per lane l (nl=l%16, s=l/16) element jj:
//   B[k_local = 16*s + jj][n = 16*j + nl]  with  c = 32*t + 16*s + jj
// Stored flat: wt[((frag*32 + lane)*16) + jj]  (one k = 4096 u16 = 8192 B)
// ---------------------------------------------------------------------------
__global__ __launch_bounds__(256) void prep_w(const float* __restrict__ W,
                                              u16* __restrict__ wt, int K) {
  int tid = blockIdx.x * blockDim.x + threadIdx.x;
  int total = K * 2 * 4 * 32 * 16;
  if (tid >= total) return;
  int jj   = tid & 15;
  int lane = (tid >> 4) & 31;
  int j    = (tid >> 9) & 3;
  int t    = (tid >> 11) & 1;
  int k    = tid >> 12;
  int nl = lane & 15, s = lane >> 4;
  int c = 32 * t + 16 * s + jj;
  int d = 16 * j + nl;
  wt[tid] = f2bf(W[((long)k * C_DIM + c) * C_DIM + d]);
}

// ---------------------------------------------------------------------------
// Async global -> LDS weight staging (CDNA5 ASYNCcnt path).
// Each wave copies its 1KB slice of the 8KB W[k] fragment block: 2 x b128/lane.
// GVS mode: mem = SADDR(64b, uniform) + VADDR(32b, per-lane) + IOFFSET.
// Same IOFFSET is applied to both the LDS and global addresses.
// ---------------------------------------------------------------------------
__device__ __forceinline__ void async_wload(const u16* __restrict__ wt,
                                            unsigned ldsoff, unsigned goff) {
  asm volatile("global_load_async_to_lds_b128 %0, %1, %2\n\t"
               "global_load_async_to_lds_b128 %0, %1, %2 offset:512"
               :: "v"(ldsoff), "v"(goff), "s"(wt) : "memory");
}

__device__ __forceinline__ void wait_async0() {
  asm volatile("s_wait_asynccnt 0x0" ::: "memory");
}

// ---------------------------------------------------------------------------
// Sparse conv via bf16 WMMA. One wave -> 32 output rows (2 tiles) x 64 cols.
// Block = 8 waves = 256 rows. W[k] double-buffered in LDS via async loads.
// ---------------------------------------------------------------------------
__global__ __launch_bounds__(256) void conv_wmma(
    const u16* __restrict__ tin, const u16* __restrict__ wt,
    const int* __restrict__ idx, const float* __restrict__ mask,
    float* __restrict__ out, int N, int K) {
  __shared__ u16 smem[2][4096];          // 2 x 8KB weight buffers

  const int tid  = threadIdx.x;
  const int lane = tid & 31;
  const int wv   = tid >> 5;             // wave in block, 0..7
  const int ml   = lane & 15;
  const int s    = lane >> 4;

  const int base0 = blockIdx.x * 256 + wv * 32;  // rows of tile 0
  const int base1 = base0 + 16;                  // rows of tile 1

  const int r0 = base0 + ml, r1 = base1 + ml;
  const int ok0 = (r0 < N), ok1 = (r1 < N);
  const int r0c = ok0 ? r0 : 0;
  const int r1c = ok1 ? r1 : 0;

  const unsigned smbase  = (unsigned)(size_t)(&smem[0][0]);
  const unsigned slice   = (unsigned)(wv * 1024 + lane * 16);

  v8f acc0[4], acc1[4];
  #pragma unroll
  for (int j = 0; j < 4; ++j)
    #pragma unroll
    for (int e = 0; e < 8; ++e) { acc0[j][e] = 0.0f; acc1[j][e] = 0.0f; }

  // prologue: stage W[0] into buffer 0
  async_wload(wt, smbase + slice, slice);
  wait_async0();
  __syncthreads();

  for (int k = 0; k < K; ++k) {
    if (k + 1 < K)   // prefetch next W into the other buffer (overlaps compute)
      async_wload(wt, smbase + (unsigned)(((k + 1) & 1) * 8192) + slice,
                  (unsigned)((k + 1) * 8192) + slice);

    // ---- gather A fragments (2 row tiles x 2 K-chunks), mask via AND ----
    const long kb = (long)k * N;
    int   n0 = idx[kb + r0c];
    int   n1 = idx[kb + r1c];
    float m0 = ok0 ? mask[kb + r0c] : 0.0f;
    float m1 = ok1 ? mask[kb + r1c] : 0.0f;
    unsigned sel0 = (m0 != 0.0f) ? 0xFFFFFFFFu : 0u;
    unsigned sel1 = (m1 != 0.0f) ? 0xFFFFFFFFu : 0u;

    const uint4* p0 = (const uint4*)(tin + (long)n0 * C_DIM);
    const uint4* p1 = (const uint4*)(tin + (long)n1 * C_DIM);
    FragA a00, a01, a10, a11;
    a00.q[0] = p0[2 * s];     a00.q[1] = p0[2 * s + 1];
    a01.q[0] = p0[2 * s + 4]; a01.q[1] = p0[2 * s + 5];
    a10.q[0] = p1[2 * s];     a10.q[1] = p1[2 * s + 1];
    a11.q[0] = p1[2 * s + 4]; a11.q[1] = p1[2 * s + 5];
    #pragma unroll
    for (int q = 0; q < 2; ++q) {
      a00.q[q].x &= sel0; a00.q[q].y &= sel0; a00.q[q].z &= sel0; a00.q[q].w &= sel0;
      a01.q[q].x &= sel0; a01.q[q].y &= sel0; a01.q[q].z &= sel0; a01.q[q].w &= sel0;
      a10.q[q].x &= sel1; a10.q[q].y &= sel1; a10.q[q].z &= sel1; a10.q[q].w &= sel1;
      a11.q[q].x &= sel1; a11.q[q].y &= sel1; a11.q[q].z &= sel1; a11.q[q].w &= sel1;
    }

    // ---- B fragments from LDS, shared by both row tiles ----
    const uint4* bq = (const uint4*)(&smem[k & 1][0]);
    #pragma unroll
    for (int j = 0; j < 4; ++j) {
      FragA b0, b1;
      const int f0 = (0 * 4 + j) * 64 + lane * 2;
      const int f1 = (1 * 4 + j) * 64 + lane * 2;
      b0.q[0] = bq[f0]; b0.q[1] = bq[f0 + 1];
      b1.q[0] = bq[f1]; b1.q[1] = bq[f1 + 1];
      acc0[j] = __builtin_amdgcn_wmma_f32_16x16x32_bf16(
          false, a00.v, false, b0.v, (short)0, acc0[j], false, false);
      acc0[j] = __builtin_amdgcn_wmma_f32_16x16x32_bf16(
          false, a01.v, false, b1.v, (short)0, acc0[j], false, false);
      acc1[j] = __builtin_amdgcn_wmma_f32_16x16x32_bf16(
          false, a10.v, false, b0.v, (short)0, acc1[j], false, false);
      acc1[j] = __builtin_amdgcn_wmma_f32_16x16x32_bf16(
          false, a11.v, false, b1.v, (short)0, acc1[j], false, false);
    }

    if (k + 1 < K) wait_async0();   // next buffer fully in LDS
    __syncthreads();                // all waves done reading current buffer
  }

  // C/D layout: lane l -> col = 16j + (l%16); vgpr r -> row = base + 8*(l/16)+r
  #pragma unroll
  for (int j = 0; j < 4; ++j)
    #pragma unroll
    for (int r = 0; r < 8; ++r) {
      int o0 = base0 + 8 * s + r;
      int o1 = base1 + 8 * s + r;
      if (o0 < N) out[(long)o0 * C_DIM + 16 * j + ml] = acc0[j][r];
      if (o1 < N) out[(long)o1 * C_DIM + 16 * j + ml] = acc1[j][r];
    }
}

// ---------------------------------------------------------------------------
// BN stats: deterministic two-stage per-channel sum / sumsq reduction
// ---------------------------------------------------------------------------
__global__ __launch_bounds__(256) void stats_partial(const float* __restrict__ x,
                                                     float* __restrict__ part, int N) {
  __shared__ float sh0[256], sh1[256];
  const int c = threadIdx.x & 63;
  const int g = threadIdx.x >> 6;
  float s0 = 0.f, s1 = 0.f;
  for (int n = blockIdx.x * 4 + g; n < N; n += gridDim.x * 4) {
    float v = x[(long)n * C_DIM + c];
    s0 += v; s1 += v * v;
  }
  sh0[threadIdx.x] = s0; sh1[threadIdx.x] = s1;
  __syncthreads();
  if (threadIdx.x < 64) {
    float a = sh0[c] + sh0[c + 64] + sh0[c + 128] + sh0[c + 192];
    float b = sh1[c] + sh1[c + 64] + sh1[c + 128] + sh1[c + 192];
    part[blockIdx.x * 128 + c]      = a;
    part[blockIdx.x * 128 + 64 + c] = b;
  }
}

__global__ void stats_reduce(const float* __restrict__ part,
                             float* __restrict__ stats, int nblk) {
  int t = threadIdx.x;  // 128 threads: [0..63]=sum, [64..127]=sumsq
  float s = 0.f;
  for (int b = 0; b < nblk; ++b) s += part[b * 128 + t];
  stats[t] = s;
}

// ---------------------------------------------------------------------------
// BN1 + ReLU -> bf16 mid table
// ---------------------------------------------------------------------------
__global__ __launch_bounds__(256) void bn_relu_bf16(
    const float* __restrict__ x, const float* __restrict__ stats,
    const float* __restrict__ gamma, const float* __restrict__ beta,
    u16* __restrict__ t1, int N) {
  long i = (long)blockIdx.x * blockDim.x + threadIdx.x;
  if (i >= (long)N * C_DIM) return;
  int c = (int)(i & 63);
  float invN = 1.0f / (float)N;
  float mean = stats[c] * invN;
  float var  = stats[64 + c] * invN - mean * mean;
  float sc = gamma[c] * rsqrtf(var + EPSV);
  float bi = beta[c] - mean * sc;
  float y = fmaxf(x[i] * sc + bi, 0.0f);
  t1[i] = f2bf(y);
}

// ---------------------------------------------------------------------------
// BN2 + residual + ReLU (in-place on d_out)
// ---------------------------------------------------------------------------
__global__ __launch_bounds__(256) void bn_res_relu(
    float* __restrict__ out, const float* __restrict__ stats,
    const float* __restrict__ gamma, const float* __restrict__ beta,
    const float* __restrict__ feats, int N) {
  long i = (long)blockIdx.x * blockDim.x + threadIdx.x;
  if (i >= (long)N * C_DIM) return;
  int c = (int)(i & 63);
  float invN = 1.0f / (float)N;
  float mean = stats[c] * invN;
  float var  = stats[64 + c] * invN - mean * mean;
  float sc = gamma[c] * rsqrtf(var + EPSV);
  float bi = beta[c] - mean * sc;
  float y = out[i] * sc + bi + feats[i];
  out[i] = fmaxf(y, 0.0f);
}

// ---------------------------------------------------------------------------
extern "C" void kernel_launch(void* const* d_in, const int* in_sizes, int n_in,
                              void* d_out, int out_size, void* d_ws, size_t ws_size,
                              hipStream_t stream) {
  const float* feats  = (const float*)d_in[0];
  const float* W1     = (const float*)d_in[1];
  const float* gamma1 = (const float*)d_in[2];
  const float* beta1  = (const float*)d_in[3];
  const float* W2     = (const float*)d_in[4];
  const float* gamma2 = (const float*)d_in[5];
  const float* beta2  = (const float*)d_in[6];
  const int*   idx1   = (const int*)d_in[7];
  const float* mask1  = (const float*)d_in[8];
  const int*   idx2   = (const int*)d_in[9];
  const float* mask2  = (const float*)d_in[10];
  float* outp = (float*)d_out;

  const int N = in_sizes[0] / C_DIM;
  const int K = in_sizes[1] / (C_DIM * C_DIM);
  const long NC = (long)N * C_DIM;

  // carve workspace (256B-aligned chunks)
  size_t off = 0;
  char* wsb = (char*)d_ws;
  auto carve = [&](size_t bytes) -> char* {
    char* p = wsb + off;
    off += (bytes + 255) & ~((size_t)255);
    return p;
  };
  u16*   t0    = (u16*)carve(NC * sizeof(u16));
  u16*   t1    = (u16*)carve(NC * sizeof(u16));
  float* out1  = (float*)carve(NC * sizeof(float));
  u16*   wt1   = (u16*)carve((size_t)K * 4096 * sizeof(u16));
  u16*   wt2   = (u16*)carve((size_t)K * 4096 * sizeof(u16));
  float* part  = (float*)carve((size_t)PBLK * 128 * sizeof(float));
  float* stat1 = (float*)carve(512);
  float* stat2 = (float*)carve(512);
  (void)ws_size; (void)n_in; (void)out_size;

  const int TB = 256;
  const int ecnt_blocks = (int)((NC + TB - 1) / TB);
  const int wblk = (K * 4096 + TB - 1) / TB;
  const int conv_blocks = (N + 255) / 256;   // 8 waves/block, 32 rows/wave

  // 1) bf16 feature table + swizzled bf16 weight fragments
  cvt_bf16<<<ecnt_blocks, TB, 0, stream>>>(feats, t0, NC);
  prep_w<<<wblk, TB, 0, stream>>>(W1, wt1, K);
  prep_w<<<wblk, TB, 0, stream>>>(W2, wt2, K);

  // 2) conv1 (WMMA)
  conv_wmma<<<conv_blocks, TB, 0, stream>>>(t0, wt1, idx1, mask1, out1, N, K);

  // 3) BN1 stats
  stats_partial<<<PBLK, TB, 0, stream>>>(out1, part, N);
  stats_reduce<<<1, 128, 0, stream>>>(part, stat1, PBLK);

  // 4) BN1 + ReLU -> bf16 mid table
  bn_relu_bf16<<<ecnt_blocks, TB, 0, stream>>>(out1, stat1, gamma1, beta1, t1, N);

  // 5) conv2 (WMMA) -> d_out (raw)
  conv_wmma<<<conv_blocks, TB, 0, stream>>>(t1, wt2, idx2, mask2, outp, N, K);

  // 6) BN2 stats
  stats_partial<<<PBLK, TB, 0, stream>>>(outp, part, N);
  stats_reduce<<<1, 128, 0, stream>>>(part, stat2, PBLK);

  // 7) BN2 + residual + ReLU in-place
  bn_res_relu<<<ecnt_blocks, TB, 0, stream>>>(outp, stat2, gamma2, beta2, feats, N);
}
